// LSTM_50328426774676
// MI455X (gfx1250) — matrix operand order repaired and checked by reading, compiled, not run on hardware
//
#include <hip/hip_runtime.h>
#include <hip/hip_bf16.h>
#include <math.h>

#define T_STEPS 4096
#define IN_DIM  1024
#define H_DIM   2048
#define OUT_DIM 512
#define NWG     64
#define LDSPAD  2056   // 2048 + 8 bf16 pad: row stride 1028 dwords -> 4-bank skew
#define CHUNK_K 128    // async-staged K-chunk for streamed waves
#define NCHUNK  (H_DIM / CHUNK_K)   // 16
#define SROW    136    // staged row stride (bf16): 68 dwords -> 4-bank skew

typedef __attribute__((ext_vector_type(16))) __bf16 v16bf;
typedef __attribute__((ext_vector_type(8)))  __bf16 bf8v;
typedef __attribute__((ext_vector_type(8)))  float  v8f;

// ---------- WMMA fragment loaders (wave32, bf16, 16x16x32) ----------
__device__ __forceinline__ v16bf load_a_frag(const __bf16* base, int ld, int lane) {
  int m = lane & 15, hf = lane >> 4;
  const __bf16* p = base + (size_t)m * ld + hf * 16;
  bf8v lo = *(const bf8v*)(p);
  bf8v hi = *(const bf8v*)(p + 8);
  v16bf r;
#pragma unroll
  for (int i = 0; i < 8; ++i) { r[i] = lo[i]; r[i + 8] = hi[i]; }
  return r;
}

__device__ __forceinline__ v16bf load_b_frag(const __bf16* base, int ld, int lane) {
  int n = lane & 15, hf = lane >> 4;
  const __bf16* p = base + (size_t)n * ld + hf * 16;
  bf8v lo = *(const bf8v*)(p);
  bf8v hi = *(const bf8v*)(p + 8);
  v16bf r;
#pragma unroll
  for (int i = 0; i < 8; ++i) { r[i] = lo[i]; r[i + 8] = hi[i]; }
  return r;
}

__device__ __forceinline__ v16bf load_bvec_frag(const __bf16* vec, int lane) {
  int hf = lane >> 4;
  const __bf16* p = vec + hf * 16;
  bf8v lo = *(const bf8v*)(p);
  bf8v hi = *(const bf8v*)(p + 8);
  v16bf r;
#pragma unroll
  for (int i = 0; i < 8; ++i) { r[i] = lo[i]; r[i + 8] = hi[i]; }
  return r;
}

// ---------- CDNA5 async global->LDS copy (ASYNCcnt-tracked) ----------
__device__ __forceinline__ void async_ld_b128(unsigned lds_off, const void* gaddr) {
  asm volatile("global_load_async_to_lds_b128 %0, %1, off"
               :: "v"(lds_off), "v"(gaddr) : "memory");
}
__device__ __forceinline__ void wait_async_le16() {
  asm volatile("s_wait_asynccnt 0x10" ::: "memory");
}
__device__ __forceinline__ void wait_async_le8() {
  asm volatile("s_wait_asynccnt 0x8" ::: "memory");
}
__device__ __forceinline__ void wait_async_0() {
  asm volatile("s_wait_asynccnt 0x0" ::: "memory");
}

// issue one 16x128 bf16 chunk: 8 lane-scattered async b128 loads
__device__ __forceinline__ void issue_chunk(const __bf16* Wrow, int k0,
                                            __bf16* stage_wave, int lane) {
#pragma unroll
  for (int q = 0; q < 8; ++q) {
    int e  = q * 32 + lane;        // 16B granule index
    int r  = e >> 4;               // row 0..15
    int c8 = e & 15;               // 8-bf16 column chunk
    const __bf16* g = Wrow + (size_t)r * H_DIM + k0 + c8 * 8;
    unsigned loff = (unsigned)(uintptr_t)(stage_wave + r * SROW + c8 * 8);
    async_ld_b128(loff, (const void*)g);
  }
}

// ---------- prep kernels ----------
__global__ void cvt_bf16(const float* __restrict__ src, __bf16* __restrict__ dst,
                         int rows, int cols, int ld, int coff) {
  size_t total = (size_t)rows * cols;
  for (size_t i = (size_t)blockIdx.x * blockDim.x + threadIdx.x; i < total;
       i += (size_t)gridDim.x * blockDim.x) {
    size_t r = i / cols, c = i % cols;
    dst[i] = (__bf16)src[r * ld + coff + c];
  }
}

__global__ void init_state(float* __restrict__ b_all,
                           const float* __restrict__ bi, const float* __restrict__ bfv,
                           const float* __restrict__ bo, const float* __restrict__ bc,
                           __bf16* __restrict__ hbf, int* __restrict__ flags) {
  int i = blockIdx.x * blockDim.x + threadIdx.x;           // 0..8191
  if (i < H_DIM) {
    b_all[0 * H_DIM + i] = bi[i];
    b_all[1 * H_DIM + i] = bfv[i];
    b_all[2 * H_DIM + i] = bo[i];
    b_all[3 * H_DIM + i] = bc[i];
  }
  if (i < 2 * H_DIM) hbf[i] = (__bf16)0.0f;
  if (i < T_STEPS)   flags[i] = 0;
}

// ---------- phase 1: xg[g][t][n] = x @ Wgx^T + bg ----------
__global__ void __launch_bounds__(256)
gemm_pre(const __bf16* __restrict__ xb, const __bf16* __restrict__ Wx,
         const float* __restrict__ b_all, float* __restrict__ xg) {
  int lane = threadIdx.x & 31, w = threadIdx.x >> 5;
  int g  = blockIdx.z;
  int n0 = blockIdx.x * 512 + w * 64;
  int t0 = blockIdx.y * 16;
  const __bf16* A = xb + (size_t)t0 * IN_DIM;
  const __bf16* B = Wx + ((size_t)g * H_DIM + n0) * IN_DIM;
  v8f acc[4] = {};
  v16bf a0 = load_a_frag(A, IN_DIM, lane);
  v16bf b0[4];
#pragma unroll
  for (int j = 0; j < 4; ++j) b0[j] = load_b_frag(B + (size_t)j * 16 * IN_DIM, IN_DIM, lane);
#pragma unroll 2
  for (int k0 = 0; k0 < IN_DIM - 32; k0 += 32) {
    v16bf a1 = load_a_frag(A + k0 + 32, IN_DIM, lane);
    v16bf b1[4];
#pragma unroll
    for (int j = 0; j < 4; ++j)
      b1[j] = load_b_frag(B + (size_t)j * 16 * IN_DIM + k0 + 32, IN_DIM, lane);
#pragma unroll
    for (int j = 0; j < 4; ++j)
      acc[j] = __builtin_amdgcn_wmma_f32_16x16x32_bf16(
          false, a0, false, b0[j], (short)0, acc[j], false, false);
    a0 = a1;
#pragma unroll
    for (int j = 0; j < 4; ++j) b0[j] = b1[j];
  }
#pragma unroll
  for (int j = 0; j < 4; ++j)
    acc[j] = __builtin_amdgcn_wmma_f32_16x16x32_bf16(
        false, a0, false, b0[j], (short)0, acc[j], false, false);

  int m = lane & 15, hf = lane >> 4;
#pragma unroll
  for (int j = 0; j < 4; ++j) {
    int nn = n0 + j * 16 + m;
    float bias = b_all[g * H_DIM + nn];
    float* C = xg + ((size_t)g * T_STEPS + t0) * H_DIM + nn;
#pragma unroll
    for (int v = 0; v < 8; ++v)
      C[(size_t)(v + 8 * hf) * H_DIM] = acc[j][v] + bias;
  }
}

// ---------- phase 2: persistent recurrent scan ----------
// Waves 0-3 (gates i,f): weights pinned in LDS (loaded once).
// Waves 4-7 (gates o,c): weights via async global->LDS, 2 chunks in flight.
// h vector staged in LDS once per step; all B-fragments are LDS broadcasts.
__global__ void __launch_bounds__(256)
lstm_scan(const __bf16* __restrict__ Wh, const float* __restrict__ xg,
          __bf16* __restrict__ hbf, float* __restrict__ hfin,
          float* __restrict__ cfin, int* __restrict__ flags) {
  __shared__ __align__(16) __bf16 ldsW[64 * LDSPAD];            // 263,168 B
  __shared__ __align__(16) __bf16 stage[4 * 3 * 16 * SROW];     //  52,224 B
  __shared__ __align__(16) __bf16 hstage[H_DIM];                //   4,096 B
  __shared__ float lds_y[128];
  __shared__ float lds_c[32];
  int tid = threadIdx.x, lane = tid & 31, w = tid >> 5;
  int wg = blockIdx.x;                 // owns n in [wg*32, wg*32+32)
  int g = w >> 1, s = w & 1;           // wave tile: gate g, n-subtile s
  const __bf16* Wrow = Wh + (size_t)(g * H_DIM + wg * 32 + s * 16) * H_DIM;

  if (w < 4) {                         // one-time pin of gates 0,1 rows
    __bf16* dst = ldsW + (size_t)w * 16 * LDSPAD;
    for (int idx = lane; idx < 16 * (H_DIM / 8); idx += 32) {
      int r = idx >> 8;
      int c = (idx & 255) * 8;
      *(bf8v*)(dst + (size_t)r * LDSPAD + c) =
          *(const bf8v*)(Wrow + (size_t)r * H_DIM + c);
    }
  }
  if (tid < 32) lds_c[tid] = 0.0f;
  __syncthreads();

  const __bf16* ldsBase = ldsW + (size_t)w * 16 * LDSPAD;
  int wi = (w >= 4) ? (w - 4) : 0;
  __bf16* stb = stage + (size_t)wi * 3 * 16 * SROW;   // 3 rotating buffers

  for (int t = 0; t < T_STEPS; ++t) {
    if (t > 0) {
      if (tid == 0) {
        while (__hip_atomic_load(&flags[t - 1], __ATOMIC_ACQUIRE,
                                 __HIP_MEMORY_SCOPE_AGENT) < NWG)
          __builtin_amdgcn_s_sleep(1);
      }
      __syncthreads();
    }
    const __bf16* hsrc = hbf + (size_t)(t & 1) * H_DIM;
    // streamed waves: start weight prefetch before anything else this step
    if (w >= 4) {
      issue_chunk(Wrow, 0 * CHUNK_K, stb + 0 * 16 * SROW, lane);
      issue_chunk(Wrow, 1 * CHUNK_K, stb + 1 * 16 * SROW, lane);
    }
    // all threads: stage h into LDS (256 x b128 = 4KB)
    *(bf8v*)(hstage + tid * 8) = *(const bf8v*)(hsrc + tid * 8);
    __syncthreads();

    v8f acc = {};
    if (w < 4) {
      // pure-LDS path: pinned weights + staged h
#pragma unroll 4
      for (int k0 = 0; k0 < H_DIM; k0 += 32) {
        v16bf a = load_a_frag(ldsBase + k0, LDSPAD, lane);
        v16bf b = load_bvec_frag(hstage + k0, lane);
        acc = __builtin_amdgcn_wmma_f32_16x16x32_bf16(
            false, a, false, b, (short)0, acc, false, false);
      }
    } else {
      // async-staged weights, 2 chunks in flight (triple buffer)
      for (int ch = 0; ch < NCHUNK; ++ch) {
        __bf16* cur = stb + (size_t)(ch % 3) * 16 * SROW;
        if (ch + 2 < NCHUNK) {
          issue_chunk(Wrow, (ch + 2) * CHUNK_K, stb + (size_t)((ch + 2) % 3) * 16 * SROW, lane);
          wait_async_le16();           // chunk ch landed; ch+1, ch+2 in flight
        } else if (ch + 1 < NCHUNK) {
          wait_async_le8();
        } else {
          wait_async_0();
        }
        int kb = ch * CHUNK_K;
#pragma unroll
        for (int kk = 0; kk < CHUNK_K; kk += 32) {
          v16bf a = load_a_frag(cur + kk, SROW, lane);
          v16bf b = load_bvec_frag(hstage + kb + kk, lane);
          acc = __builtin_amdgcn_wmma_f32_16x16x32_bf16(
              false, a, false, b, (short)0, acc, false, false);
        }
      }
    }
    int hf = lane >> 4;
    if ((lane & 15) == 0) {            // column n=0 lives in lanes 0 and 16
#pragma unroll
      for (int v = 0; v < 8; ++v) lds_y[w * 16 + v + 8 * hf] = acc[v];
    }
    __syncthreads();
    if (tid < 32) {
      int n  = wg * 32 + tid;
      int sl = tid >> 4, m = tid & 15;
      float yi = lds_y[(0 * 2 + sl) * 16 + m];
      float yf = lds_y[(1 * 2 + sl) * 16 + m];
      float yo = lds_y[(2 * 2 + sl) * 16 + m];
      float yc = lds_y[(3 * 2 + sl) * 16 + m];
      size_t base = (size_t)t * H_DIM + n;
      float pi = xg[base] + yi;
      float pf = xg[(size_t)1 * T_STEPS * H_DIM + base] + yf;
      float po = xg[(size_t)2 * T_STEPS * H_DIM + base] + yo;
      float pc = xg[(size_t)3 * T_STEPS * H_DIM + base] + yc;
      float ig = 1.0f / (1.0f + __expf(-pi));
      float fg = 1.0f / (1.0f + __expf(-pf));
      float og = 1.0f / (1.0f + __expf(-po));
      float ct = tanhf(pc);
      float c  = fg * lds_c[tid] + ig * ct;
      float h  = og * tanhf(c);
      lds_c[tid] = c;
      hbf[(size_t)((t + 1) & 1) * H_DIM + n] = (__bf16)h;
      if (t == T_STEPS - 1) { hfin[n] = h; cfin[n] = c; }
      __threadfence();
    }
    __syncthreads();
    if (tid == 0)
      __hip_atomic_fetch_add(&flags[t], 1, __ATOMIC_RELEASE,
                             __HIP_MEMORY_SCOPE_AGENT);
  }
}

// ---------- phase 3: out = W2 @ cat(h,c) + b2 ; also emit h ----------
__global__ void __launch_bounds__(256)
head_kernel(const float* __restrict__ W2, const float* __restrict__ b2,
            const float* __restrict__ hfin, const float* __restrict__ cfin,
            float* __restrict__ out) {
  int bx = blockIdx.x;
  if (bx < 64) {
    int w = threadIdx.x >> 5, lane = threadIdx.x & 31;
    int r = bx * 8 + w;
    const float* row = W2 + (size_t)r * (2 * H_DIM);
    float sum = 0.0f;
    for (int k = lane; k < H_DIM; k += 32) sum += row[k] * hfin[k];
    for (int k = lane; k < H_DIM; k += 32) sum += row[H_DIM + k] * cfin[k];
#pragma unroll
    for (int off = 16; off > 0; off >>= 1) sum += __shfl_xor(sum, off, 32);
    if (lane == 0) out[r] = sum + b2[r];
  } else {
    int i = (bx - 64) * 256 + threadIdx.x;
    if (i < H_DIM) out[OUT_DIM + i] = hfin[i];
  }
}

// ---------- host ----------
static inline size_t align256(size_t x) { return (x + 255) & ~(size_t)255; }

extern "C" void kernel_launch(void* const* d_in, const int* in_sizes, int n_in,
                              void* d_out, int out_size, void* d_ws, size_t ws_size,
                              hipStream_t stream) {
  const float* x   = (const float*)d_in[0];
  const float* Wi  = (const float*)d_in[1];
  const float* bi  = (const float*)d_in[2];
  const float* Wf_ = (const float*)d_in[3];
  const float* bfv = (const float*)d_in[4];
  const float* Wo  = (const float*)d_in[5];
  const float* bo  = (const float*)d_in[6];
  const float* Wc  = (const float*)d_in[7];
  const float* bc  = (const float*)d_in[8];
  const float* W2  = (const float*)d_in[9];
  const float* b2  = (const float*)d_in[10];

  char* ws = (char*)d_ws;
  size_t off = 0;
  float*  xg    = (float*)(ws + off);  off = align256(off + (size_t)4 * T_STEPS * H_DIM * 4);
  __bf16* xb    = (__bf16*)(ws + off); off = align256(off + (size_t)T_STEPS * IN_DIM * 2);
  __bf16* Wx    = (__bf16*)(ws + off); off = align256(off + (size_t)4 * H_DIM * IN_DIM * 2);
  __bf16* Wh    = (__bf16*)(ws + off); off = align256(off + (size_t)4 * H_DIM * H_DIM * 2);
  __bf16* hbf   = (__bf16*)(ws + off); off = align256(off + (size_t)2 * H_DIM * 2);
  float*  b_all = (float*)(ws + off);  off = align256(off + (size_t)4 * H_DIM * 4);
  float*  hfin  = (float*)(ws + off);  off = align256(off + (size_t)H_DIM * 4);
  float*  cfin  = (float*)(ws + off);  off = align256(off + (size_t)H_DIM * 4);
  int*    flags = (int*)(ws + off);    off = align256(off + (size_t)T_STEPS * 4);

  init_state<<<32, 256, 0, stream>>>(b_all, bi, bfv, bo, bc, hbf, flags);

  cvt_bf16<<<256, 256, 0, stream>>>(x, xb, T_STEPS, IN_DIM, IN_DIM, 0);
  const float* Wg[4] = {Wi, Wf_, Wo, Wc};
  for (int g = 0; g < 4; ++g) {
    cvt_bf16<<<256, 256, 0, stream>>>(Wg[g], Wx + (size_t)g * H_DIM * IN_DIM,
                                      H_DIM, IN_DIM, IN_DIM + H_DIM, 0);
    cvt_bf16<<<256, 256, 0, stream>>>(Wg[g], Wh + (size_t)g * H_DIM * H_DIM,
                                      H_DIM, H_DIM, IN_DIM + H_DIM, IN_DIM);
  }

  gemm_pre<<<dim3(H_DIM / 512, T_STEPS / 16, 4), 256, 0, stream>>>(xb, Wx, b_all, xg);
  lstm_scan<<<NWG, 256, 0, stream>>>(Wh, xg, hbf, hfin, cfin, flags);
  head_kernel<<<72, 256, 0, stream>>>(W2, b2, hfin, cfin, (float*)d_out);
}
